// MoELayer_57664230916132
// MI455X (gfx1250) — matrix operand order repaired and checked by reading, not compiled, not run on hardware
//
#include <hip/hip_runtime.h>

// Problem constants (from reference): B=4,S=2048 -> 8192 tokens
#define TOKS   8192
#define DMODEL 1024
#define NEXP   8
#define HDIM   256
#define NOUT   1024
#define KTWO   (NEXP * HDIM)   // 2048: flattened (e,h) reduction axis

typedef __attribute__((ext_vector_type(16))) __bf16 v16bf;
typedef __attribute__((ext_vector_type(8)))  float  v8f;

union FragBF {
    unsigned  u[8];
    uint4     q[2];
    v16bf     v;
};

__device__ __forceinline__ unsigned short f2bf(float f) {
    union { float f; unsigned u; } c; c.f = f;
    unsigned u = c.u;
    u += 0x7fffu + ((u >> 16) & 1u);   // round-to-nearest-even
    return (unsigned short)(u >> 16);
}

// ---- CDNA5 async global->LDS copy (ASYNCcnt-tracked), ISA 15.18.3 op 98 ----
// INST_OFFSET is added to BOTH the LDS and global address (ISA 8.4.4), and our
// LDS/global chunk offsets advance in lockstep -> put them in the immediate.
#define ASYNC_LD_B128(ldsoff, gaddr, OFF)                                  \
    asm volatile("global_load_async_to_lds_b128 %0, %1, off offset:" OFF   \
                 :: "v"(ldsoff), "v"(gaddr) : "memory")

__device__ __forceinline__ void wait_async0() {
    asm volatile("s_wait_asynccnt 0x0" ::: "memory");
}

// one thread stages 64B of A and 64B of B per K64 stage (8 async b128 ops)
__device__ __forceinline__ void stage_issue(const unsigned short* gA, unsigned ldsA,
                                            const unsigned short* gB, unsigned ldsB) {
    unsigned long long a = (unsigned long long)(uintptr_t)gA;
    unsigned long long b = (unsigned long long)(uintptr_t)gB;
    ASYNC_LD_B128(ldsA, a, "0");
    ASYNC_LD_B128(ldsA, a, "16");
    ASYNC_LD_B128(ldsA, a, "32");
    ASYNC_LD_B128(ldsA, a, "48");
    ASYNC_LD_B128(ldsB, b, "0");
    ASYNC_LD_B128(ldsB, b, "16");
    ASYNC_LD_B128(ldsB, b, "32");
    ASYNC_LD_B128(ldsB, b, "48");
}

// ---------- fp32 -> bf16 conversion (coalesced) ----------
__global__ __launch_bounds__(256) void k_cvt_x(const float* __restrict__ x,
                                               unsigned short* __restrict__ xb) {
    int i = (blockIdx.x * 256 + threadIdx.x) * 4;
    float4 v = *(const float4*)(x + i);
    uint2 p;
    p.x = (unsigned)f2bf(v.x) | ((unsigned)f2bf(v.y) << 16);
    p.y = (unsigned)f2bf(v.z) | ((unsigned)f2bf(v.w) << 16);
    *(uint2*)(xb + i) = p;
}

// ---------- LDS-tiled transpose: src f32 [R][C] -> dst bf16 [C][R] ----------
// grid: (R/32, C/32, nMatrices); block 256
__global__ __launch_bounds__(256) void k_transpose_bf16(const float* __restrict__ src,
                                                        unsigned short* __restrict__ dst,
                                                        int R, int C) {
    __shared__ unsigned short tile[32][33];
    const float* s = src + (size_t)blockIdx.z * R * C;
    unsigned short* d = dst + (size_t)blockIdx.z * R * C;
    const int r0 = blockIdx.x * 32, c0 = blockIdx.y * 32;
    const int tr = threadIdx.x >> 5;   // 0..7
    const int tc = threadIdx.x & 31;   // 0..31
#pragma unroll
    for (int j = 0; j < 4; ++j) {
        int r = tr + j * 8;
        tile[r][tc] = f2bf(s[(size_t)(r0 + r) * C + c0 + tc]);
    }
    __syncthreads();
#pragma unroll
    for (int j = 0; j < 4; ++j) {
        int c = tr + j * 8;            // output row = source column
        d[(size_t)(c0 + c) * R + r0 + tc] = tile[tc][c];
    }
}

// ---------- gating: logits -> softmax -> sort descending ----------
__global__ __launch_bounds__(256) void k_gate(const float* __restrict__ x,
                                              const float* __restrict__ wg,
                                              const float* __restrict__ bg,
                                              float* __restrict__ probs) {
    int t = blockIdx.x * 256 + threadIdx.x;
    float acc[NEXP];
#pragma unroll
    for (int e = 0; e < NEXP; ++e) acc[e] = bg[e];
    const float* xr = x + t * DMODEL;
    for (int d = 0; d < DMODEL; d += 4) {
        float4 xv = *(const float4*)(xr + d);
        const float* wr = wg + d * NEXP;
#pragma unroll
        for (int e = 0; e < NEXP; ++e) {
            acc[e] = fmaf(xv.x, wr[e], acc[e]);
            acc[e] = fmaf(xv.y, wr[NEXP + e], acc[e]);
            acc[e] = fmaf(xv.z, wr[2 * NEXP + e], acc[e]);
            acc[e] = fmaf(xv.w, wr[3 * NEXP + e], acc[e]);
        }
    }
    float mx = acc[0];
#pragma unroll
    for (int e = 1; e < NEXP; ++e) mx = fmaxf(mx, acc[e]);
    float s = 0.f;
#pragma unroll
    for (int e = 0; e < NEXP; ++e) { acc[e] = __expf(acc[e] - mx); s += acc[e]; }
    float inv = 1.f / s;
#pragma unroll
    for (int e = 0; e < NEXP; ++e) acc[e] *= inv;
    // fully-unrolled bubble network: constant indices -> stays in VGPRs
#pragma unroll
    for (int i = 0; i < NEXP; ++i)
#pragma unroll
        for (int j = 0; j < NEXP - 1 - i; ++j) {
            float a = acc[j], b = acc[j + 1];
            acc[j]     = fmaxf(a, b);
            acc[j + 1] = fminf(a, b);
        }
#pragma unroll
    for (int e = 0; e < NEXP; ++e) probs[t * NEXP + e] = acc[e];
}

// ---------- WMMA fragment loaders (LDS, bf16), row stride 64 elements ----------
// A (16x32): lane m=lane&15; half=lane>>4; K pairs {half*8+2p} and {16+half*8+2p}
__device__ __forceinline__ v16bf load_fragA(const unsigned short* tile, int row, int lane) {
    const unsigned short* r = tile + row * 64 + (lane >> 4) * 8;
    FragBF f;
#pragma unroll
    for (int p = 0; p < 4; ++p) f.u[p]     = *(const unsigned*)(r + 2 * p);
#pragma unroll
    for (int p = 0; p < 4; ++p) f.u[4 + p] = *(const unsigned*)(r + 16 + 2 * p);
    return f.v;
}
// B (32x16, stored transposed [n][k]): half=lane>>4; K {half*8..+7} and {16+half*8..+7}
__device__ __forceinline__ v16bf load_fragB(const unsigned short* tile, int row, int lane) {
    const unsigned short* r = tile + row * 64 + (lane >> 4) * 8;
    FragBF f;
    f.q[0] = *(const uint4*)(r);
    f.q[1] = *(const uint4*)(r + 16);
    return f.v;
}

#define WMMA_BF16(A, B, C) \
    __builtin_amdgcn_wmma_f32_16x16x32_bf16(false, (A), false, (B), (short)0, (C), false, false)

// wave micro-kernel: load all 6 fragments first (distinct regs), then 8 WMMAs
__device__ __forceinline__ void mma_step(const unsigned short* Ab, const unsigned short* Bb,
                                         int kk, int mw, int nw, int lrow, int lane,
                                         v8f acc[2][4]) {
    v16bf a0 = load_fragA(Ab + kk, mw + lrow, lane);
    v16bf a1 = load_fragA(Ab + kk, mw + 16 + lrow, lane);
    v16bf b0 = load_fragB(Bb + kk, nw + lrow, lane);
    v16bf b1 = load_fragB(Bb + kk, nw + 16 + lrow, lane);
    v16bf b2 = load_fragB(Bb + kk, nw + 32 + lrow, lane);
    v16bf b3 = load_fragB(Bb + kk, nw + 48 + lrow, lane);
    acc[0][0] = WMMA_BF16(a0, b0, acc[0][0]);
    acc[1][0] = WMMA_BF16(a1, b0, acc[1][0]);
    acc[0][1] = WMMA_BF16(a0, b1, acc[0][1]);
    acc[1][1] = WMMA_BF16(a1, b1, acc[1][1]);
    acc[0][2] = WMMA_BF16(a0, b2, acc[0][2]);
    acc[1][2] = WMMA_BF16(a1, b2, acc[1][2]);
    acc[0][3] = WMMA_BF16(a0, b3, acc[0][3]);
    acc[1][3] = WMMA_BF16(a1, b3, acc[1][3]);
}

// ---------- Phase A: HW[t, e*H+h] = p_sorted[t,e] * gelu(X@W1[e] + b1[e]) ----------
__global__ __launch_bounds__(256) void k_ffn1(const unsigned short* __restrict__ xb,
                                              const unsigned short* __restrict__ w1t,
                                              const float* __restrict__ b1,
                                              const float* __restrict__ probs,
                                              unsigned short* __restrict__ hw) {
    __shared__ unsigned short As[2][128 * 64];   // 2 x 16 KB
    __shared__ unsigned short Bs[2][128 * 64];   // 2 x 16 KB
    const int tid  = threadIdx.x;
    const int wave = tid >> 5, lane = tid & 31;
    const int t0 = blockIdx.x * 128;
    const int e  = blockIdx.y;
    const int n0 = blockIdx.z * 128;
    const int mw = (wave >> 1) * 32;     // 4 m-slots
    const int nw = (wave & 1) * 64;      // 2 n-slots of 64
    const int lrow = lane & 15, lhalf = lane >> 4;

    // async staging assignment: thread -> (row, 32-elem half of K64 stage)
    const int crow = tid >> 1, chalf = tid & 1;
    const unsigned short* gA = xb + (t0 + crow) * DMODEL + chalf * 32;
    const unsigned short* gB = w1t + (e * HDIM + n0 + crow) * DMODEL + chalf * 32;
    const unsigned loff = (unsigned)(crow * 64 + chalf * 32) * 2;
    const unsigned ldsA[2] = { (unsigned)(uintptr_t)&As[0][0] + loff, (unsigned)(uintptr_t)&As[1][0] + loff };
    const unsigned ldsB[2] = { (unsigned)(uintptr_t)&Bs[0][0] + loff, (unsigned)(uintptr_t)&Bs[1][0] + loff };

    const v8f zero = {0.f,0.f,0.f,0.f,0.f,0.f,0.f,0.f};
    v8f acc[2][4];
#pragma unroll
    for (int mi = 0; mi < 2; ++mi)
#pragma unroll
        for (int ni = 0; ni < 4; ++ni) acc[mi][ni] = zero;

    const int NS = DMODEL / 64;
    stage_issue(gA, ldsA[0], gB, ldsB[0]);
    for (int s = 0; s < NS; ++s) {
        wait_async0();          // stage s landed in LDS (this wave's ops)
        __syncthreads();        // all waves landed; all done reading buf[(s+1)&1]
        if (s + 1 < NS) {
            gA += 64; gB += 64;
            stage_issue(gA, ldsA[(s + 1) & 1], gB, ldsB[(s + 1) & 1]);
        }
        const unsigned short* Ab = &As[s & 1][0];
        const unsigned short* Bb = &Bs[s & 1][0];
        mma_step(Ab, Bb, 0,  mw, nw, lrow, lane, acc);
        mma_step(Ab, Bb, 32, mw, nw, lrow, lane, acc);
    }

    float b1v[4];
#pragma unroll
    for (int ni = 0; ni < 4; ++ni) b1v[ni] = b1[e * HDIM + n0 + nw + ni * 16 + lrow];
#pragma unroll
    for (int mi = 0; mi < 2; ++mi) {
#pragma unroll
        for (int v = 0; v < 8; ++v) {
            int t = t0 + mw + mi * 16 + lhalf * 8 + v;
            float w = probs[t * NEXP + e];   // sorted prob e pairs with expert e (ref quirk)
#pragma unroll
            for (int ni = 0; ni < 4; ++ni) {
                int h = n0 + nw + ni * 16 + lrow;
                float g = acc[mi][ni][v] + b1v[ni];
                g = 0.5f * g * (1.0f + erff(g * 0.70710678118654752f));
                hw[t * KTWO + e * HDIM + h] = f2bf(g * w);
            }
        }
    }
}

// ---------- Phase B: OUT = HW[8192x2048] @ W2[2048x1024] + sum_e p*b2 ----------
__global__ __launch_bounds__(256) void k_ffn2(const unsigned short* __restrict__ hw,
                                              const unsigned short* __restrict__ w2t,
                                              const float* __restrict__ b2,
                                              const float* __restrict__ probs,
                                              float* __restrict__ out) {
    __shared__ unsigned short As[2][128 * 64];
    __shared__ unsigned short Bs[2][128 * 64];
    const int tid  = threadIdx.x;
    const int wave = tid >> 5, lane = tid & 31;
    const int t0 = blockIdx.x * 128;
    const int f0 = blockIdx.y * 128;
    const int mw = (wave >> 1) * 32;
    const int nw = (wave & 1) * 64;
    const int lrow = lane & 15, lhalf = lane >> 4;

    const int crow = tid >> 1, chalf = tid & 1;
    const unsigned short* gA = hw + (t0 + crow) * KTWO + chalf * 32;
    const unsigned short* gB = w2t + (f0 + crow) * KTWO + chalf * 32;
    const unsigned loff = (unsigned)(crow * 64 + chalf * 32) * 2;
    const unsigned ldsA[2] = { (unsigned)(uintptr_t)&As[0][0] + loff, (unsigned)(uintptr_t)&As[1][0] + loff };
    const unsigned ldsB[2] = { (unsigned)(uintptr_t)&Bs[0][0] + loff, (unsigned)(uintptr_t)&Bs[1][0] + loff };

    const v8f zero = {0.f,0.f,0.f,0.f,0.f,0.f,0.f,0.f};
    v8f acc[2][4];
#pragma unroll
    for (int mi = 0; mi < 2; ++mi)
#pragma unroll
        for (int ni = 0; ni < 4; ++ni) acc[mi][ni] = zero;

    const int NS = KTWO / 64;
    stage_issue(gA, ldsA[0], gB, ldsB[0]);
    for (int s = 0; s < NS; ++s) {
        wait_async0();
        __syncthreads();
        if (s + 1 < NS) {
            gA += 64; gB += 64;
            stage_issue(gA, ldsA[(s + 1) & 1], gB, ldsB[(s + 1) & 1]);
        }
        const unsigned short* Ab = &As[s & 1][0];
        const unsigned short* Bb = &Bs[s & 1][0];
        mma_step(Ab, Bb, 0,  mw, nw, lrow, lane, acc);
        mma_step(Ab, Bb, 32, mw, nw, lrow, lane, acc);
    }

    float b2v[4][NEXP];
#pragma unroll
    for (int ni = 0; ni < 4; ++ni) {
        int f = f0 + nw + ni * 16 + lrow;
#pragma unroll
        for (int e = 0; e < NEXP; ++e) b2v[ni][e] = b2[e * NOUT + f];
    }
#pragma unroll
    for (int mi = 0; mi < 2; ++mi) {
#pragma unroll
        for (int v = 0; v < 8; ++v) {
            int t = t0 + mw + mi * 16 + lhalf * 8 + v;
            const float4* pr = (const float4*)(probs + t * NEXP);
            float4 p0 = pr[0], p1 = pr[1];
            float pv[NEXP] = {p0.x, p0.y, p0.z, p0.w, p1.x, p1.y, p1.z, p1.w};
#pragma unroll
            for (int ni = 0; ni < 4; ++ni) {
                int f = f0 + nw + ni * 16 + lrow;
                float sv = acc[mi][ni][v];
#pragma unroll
                for (int e = 0; e < NEXP; ++e) sv = fmaf(pv[e], b2v[ni][e], sv);
                out[t * NOUT + f] = sv;
            }
        }
    }
}

extern "C" void kernel_launch(void* const* d_in, const int* in_sizes, int n_in,
                              void* d_out, int out_size, void* d_ws, size_t ws_size,
                              hipStream_t stream) {
    const float* x  = (const float*)d_in[0];
    const float* wg = (const float*)d_in[1];
    const float* bg = (const float*)d_in[2];
    const float* w1 = (const float*)d_in[3];
    const float* b1 = (const float*)d_in[4];
    const float* w2 = (const float*)d_in[5];
    const float* b2 = (const float*)d_in[6];
    float* out = (float*)d_out;

    char* ws = (char*)d_ws;
    float*          probs = (float*)(ws);                     //   256 KB
    unsigned short* xb    = (unsigned short*)(ws + 262144);   //    16 MB
    unsigned short* w1t   = (unsigned short*)(ws + 17039360); //     4 MB
    unsigned short* w2t   = (unsigned short*)(ws + 21233664); //     4 MB
    unsigned short* hwbuf = (unsigned short*)(ws + 25427968); //    32 MB

    k_cvt_x <<<8192, 256, 0, stream>>>(x, xb);
    // W1: 8 matrices of [D=1024][H=256] -> [H][D]
    k_transpose_bf16<<<dim3(DMODEL / 32, HDIM / 32, NEXP), 256, 0, stream>>>(w1, w1t, DMODEL, HDIM);
    // W2 flat: [K2=2048][NOUT=1024] -> [NOUT][K2]
    k_transpose_bf16<<<dim3(KTWO / 32, NOUT / 32, 1), 256, 0, stream>>>(w2, w2t, KTWO, NOUT);
    k_gate  <<<TOKS / 256, 256, 0, stream>>>(x, wg, bg, probs);
    k_ffn1  <<<dim3(TOKS / 128, NEXP, HDIM / 128), 256, 0, stream>>>(xb, w1t, b1, probs, hwbuf);
    k_ffn2  <<<dim3(TOKS / 128, NOUT / 128), 256, 0, stream>>>(hwbuf, w2t, b2, probs, out);
}
